// GCN_E_Att_71330816852159
// MI455X (gfx1250) — compile-verified, matrix operand order checked by reading, not run on hardware
//
#include <hip/hip_runtime.h>
#include <hip/hip_bf16.h>
#include <stdint.h>
#include <math.h>

// ---------------------------------------------------------------------------
// Problem constants (match reference setup_inputs)
// ---------------------------------------------------------------------------
#define NNODE 50000
#define NEDGE 800000
#define INDIM 512
#define H0D   256
#define H1D   256
#define H2D   128
#define SLOPE_GAT 0.2f
#define SLOPE     0.25f

typedef __bf16 bf16_t;
typedef bf16_t v16bf __attribute__((ext_vector_type(16)));
typedef float  v8f   __attribute__((ext_vector_type(8)));

// ---------------------------------------------------------------------------
// Helpers
// ---------------------------------------------------------------------------
__device__ __forceinline__ float leaky(float v, float s) {
    return v > 0.0f ? v : s * v;
}

// atomic float max via signed-int max / unsigned-int min trick
__device__ __forceinline__ void atomicMaxF(float* addr, float v) {
    if (v >= 0.0f) atomicMax((int*)addr, __float_as_int(v));
    else           atomicMin((unsigned int*)addr, __float_as_uint(v));
}

// Load 16 bf16 elements in the WMMA half-lane pattern:
// elements [0..7] at p, elements [8..15] at p+16 (i.e. K+16 group)
__device__ __forceinline__ v16bf load_tile16(const bf16_t* p) {
    union { uint4 u[2]; v16bf v; } t;
    t.u[0] = *reinterpret_cast<const uint4*>(p);
    t.u[1] = *reinterpret_cast<const uint4*>(p + 16);
    return t.v;
}

// ---------------------------------------------------------------------------
// Elementwise / conversion kernels
// ---------------------------------------------------------------------------
__global__ void fill_f32(float* __restrict__ p, float v, int n) {
    int i = blockIdx.x * blockDim.x + threadIdx.x;
    if (i < n) p[i] = v;
}

__global__ void f32_to_bf16_k(const float* __restrict__ in, bf16_t* __restrict__ out, int n) {
    int i = blockIdx.x * blockDim.x + threadIdx.x;
    if (i < n) out[i] = (bf16_t)in[i];
}

// Wt[n*K + k] = (bf16) W[k*N + n]   (W is KxN row-major; Wt is NxK row-major)
__global__ void transpose_to_bf16_k(const float* __restrict__ W, bf16_t* __restrict__ Wt,
                                    int K, int N) {
    int i = blockIdx.x * blockDim.x + threadIdx.x;
    if (i < K * N) {
        int k = i / N, n = i % N;
        Wt[(size_t)n * K + k] = (bf16_t)W[i];
    }
}

// out_bf16 = bf16( leaky(acc + bias[col], SLOPE) )
__global__ void bias_leaky_to_bf16_k(const float* __restrict__ acc, const float* __restrict__ bias,
                                     bf16_t* __restrict__ out, int total, int cols) {
    int i = blockIdx.x * blockDim.x + threadIdx.x;
    if (i < total) {
        int c = i % cols;
        out[i] = (bf16_t)leaky(acc[i] + bias[c], SLOPE);
    }
}

// d_out = leaky(acc + bias[col], SLOPE)  (final layer, f32 output)
__global__ void bias_leaky_f32_k(const float* __restrict__ acc, const float* __restrict__ bias,
                                 float* __restrict__ out, int total, int cols) {
    int i = blockIdx.x * blockDim.x + threadIdx.x;
    if (i < total) {
        int c = i % cols;
        out[i] = leaky(acc[i] + bias[c], SLOPE);
    }
}

// ---------------------------------------------------------------------------
// WMMA GEMM:  C[M,Nc] (f32) = A[M,K] (bf16, row-major) x Bt[Nc,K] (bf16, row-major, B^T)
// One wave computes a 16x64 strip (4 accumulators); 8 waves / block.
// Software-pipelined: next k-step's A + 4 B tiles prefetched into registers
// while the 4 back-to-back WMMAs consume the current ones.
// M tiles = M/16 (exact for M=50000); Nc % 64 == 0; K % 32 == 0 (>= 64).
// ---------------------------------------------------------------------------
__global__ __launch_bounds__(256) void gemm_bf16_wmma(const bf16_t* __restrict__ A,
                                                      const bf16_t* __restrict__ Bt,
                                                      float* __restrict__ C,
                                                      int M, int Nc, int K) {
    const int lane = threadIdx.x & 31;
    const int wave = threadIdx.x >> 5;
    const int idx  = lane & 15;   // row (A) / col (B) within tile
    const int half = lane >> 4;   // K half-group selector

    const int mtile = blockIdx.x * 8 + wave;       // wave-uniform
    if (mtile * 16 >= M) return;                   // uniform branch: EXEC stays all-1 for WMMA
    const int m0 = mtile * 16;
    const int n0 = blockIdx.y * 64;

    const bf16_t* arow = A  + (size_t)(m0 + idx) * K + half * 8;
    const bf16_t* brow = Bt + (size_t)(n0 + idx) * K + half * 8;
    const size_t bstride = (size_t)16 * K;

    v8f acc0 = {}, acc1 = {}, acc2 = {}, acc3 = {};

    // prologue: tiles for k-step 0
    v16bf a  = load_tile16(arow);
    v16bf b0 = load_tile16(brow);
    v16bf b1 = load_tile16(brow + bstride);
    v16bf b2 = load_tile16(brow + 2 * bstride);
    v16bf b3 = load_tile16(brow + 3 * bstride);

    for (int kb = 32; kb < K; kb += 32) {
        // prefetch next k-step while WMMAs run on current registers
        v16bf an  = load_tile16(arow + kb);
        v16bf bn0 = load_tile16(brow + kb);
        v16bf bn1 = load_tile16(brow + kb + bstride);
        v16bf bn2 = load_tile16(brow + kb + 2 * bstride);
        v16bf bn3 = load_tile16(brow + kb + 3 * bstride);

        acc0 = __builtin_amdgcn_wmma_f32_16x16x32_bf16(false, a, false, b0, (short)0, acc0, false, false);
        acc1 = __builtin_amdgcn_wmma_f32_16x16x32_bf16(false, a, false, b1, (short)0, acc1, false, false);
        acc2 = __builtin_amdgcn_wmma_f32_16x16x32_bf16(false, a, false, b2, (short)0, acc2, false, false);
        acc3 = __builtin_amdgcn_wmma_f32_16x16x32_bf16(false, a, false, b3, (short)0, acc3, false, false);

        a = an; b0 = bn0; b1 = bn1; b2 = bn2; b3 = bn3;
    }

    // epilogue: last k-step
    acc0 = __builtin_amdgcn_wmma_f32_16x16x32_bf16(false, a, false, b0, (short)0, acc0, false, false);
    acc1 = __builtin_amdgcn_wmma_f32_16x16x32_bf16(false, a, false, b1, (short)0, acc1, false, false);
    acc2 = __builtin_amdgcn_wmma_f32_16x16x32_bf16(false, a, false, b2, (short)0, acc2, false, false);
    acc3 = __builtin_amdgcn_wmma_f32_16x16x32_bf16(false, a, false, b3, (short)0, acc3, false, false);

    // C/D layout: lane<16 holds rows m0..m0+7 (VGPR r -> row r), lane>=16 rows m0+8..15; col = idx
    const int mrow0 = m0 + half * 8;
    v8f accs[4] = {acc0, acc1, acc2, acc3};
#pragma unroll
    for (int j = 0; j < 4; ++j) {
        const int n = n0 + j * 16 + idx;
#pragma unroll
        for (int r = 0; r < 8; ++r)
            C[(size_t)(mrow0 + r) * Nc + n] = accs[j][r];
    }
}

// ---------------------------------------------------------------------------
// GAT attention dots: a_s[n] = h[n,:].att_src ; a_d[n] = h[n,:].att_dst
// One wave32 per node, shuffle reduction.
// ---------------------------------------------------------------------------
__global__ __launch_bounds__(256) void gat_dots_k(const float* __restrict__ h,
                                                  const float* __restrict__ att_src,
                                                  const float* __restrict__ att_dst,
                                                  float* __restrict__ a_s,
                                                  float* __restrict__ a_d) {
    const int lane = threadIdx.x & 31;
    const int node = blockIdx.x * 8 + (threadIdx.x >> 5);
    if (node >= NNODE) return;
    const float* row = h + (size_t)node * H0D;
    float s1 = 0.0f, s2 = 0.0f;
#pragma unroll
    for (int it = 0; it < H0D / 32; ++it) {
        int c = lane + it * 32;
        float v = row[c];
        s1 += v * att_src[c];
        s2 += v * att_dst[c];
    }
#pragma unroll
    for (int off = 16; off > 0; off >>= 1) {
        s1 += __shfl_xor(s1, off, 32);
        s2 += __shfl_xor(s2, off, 32);
    }
    if (lane == 0) { a_s[node] = s1; a_d[node] = s2; }
}

// ---------------------------------------------------------------------------
// Edge softmax passes (E real edges + N self loops)
// ---------------------------------------------------------------------------
__global__ void gat_edge_logits_k(const int* __restrict__ src, const int* __restrict__ dst,
                                  const float* __restrict__ a_s, const float* __restrict__ a_d,
                                  float* __restrict__ e, float* __restrict__ e_max) {
    int i = blockIdx.x * blockDim.x + threadIdx.x;
    if (i >= NEDGE + NNODE) return;
    int s, d;
    if (i < NEDGE) { s = src[i]; d = dst[i]; } else { s = d = i - NEDGE; }
    float ev = leaky(a_s[s] + a_d[d], SLOPE_GAT);
    e[i] = ev;
    atomicMaxF(&e_max[d], ev);
}

__global__ void gat_edge_exp_k(const int* __restrict__ dst,
                               const float* __restrict__ e, const float* __restrict__ e_max,
                               float* __restrict__ ee, float* __restrict__ denom) {
    int i = blockIdx.x * blockDim.x + threadIdx.x;
    if (i >= NEDGE + NNODE) return;
    int d = (i < NEDGE) ? dst[i] : (i - NEDGE);
    float v = __expf(e[i] - e_max[d]);
    ee[i] = v;
    atomicAdd(&denom[d], v);
}

// out[d,:] += (ee/denom[d]) * h[s,:]  — one wave per edge, coalesced 32-lane rows
__global__ __launch_bounds__(256) void gat_aggregate_k(const int* __restrict__ src,
                                                       const int* __restrict__ dst,
                                                       const float* __restrict__ ee,
                                                       const float* __restrict__ denom,
                                                       const float* __restrict__ h,
                                                       float* __restrict__ out) {
    const int lane = threadIdx.x & 31;
    const int i = blockIdx.x * 8 + (threadIdx.x >> 5);
    if (i >= NEDGE + NNODE) return;
    int s, d;
    if (i < NEDGE) { s = src[i]; d = dst[i]; } else { s = d = i - NEDGE; }
    const float coef = ee[i] / denom[d];
    const float* hrow = h + (size_t)s * H0D;
    float* orow = out + (size_t)d * H0D;
#pragma unroll
    for (int it = 0; it < H0D / 32; ++it) {
        int c = lane + it * 32;
        atomicAdd(&orow[c], coef * hrow[c]);
    }
}

// ---------------------------------------------------------------------------
// GraphConv scatter: acc[row,:] += w * sup[col,:]  — one wave per edge
// ---------------------------------------------------------------------------
__global__ __launch_bounds__(256) void gc_scatter_k(const int* __restrict__ row,
                                                    const int* __restrict__ col,
                                                    const float* __restrict__ ew,
                                                    const float* __restrict__ sup,
                                                    float* __restrict__ acc, int cols) {
    const int lane = threadIdx.x & 31;
    const int i = blockIdx.x * 8 + (threadIdx.x >> 5);
    if (i >= NEDGE) return;
    const int r = row[i], c = col[i];
    const float w = ew[i];
    const float* srow = sup + (size_t)c * cols;
    float* arow = acc + (size_t)r * cols;
    for (int it = 0; it < cols / 32; ++it) {
        int cc = lane + it * 32;
        atomicAdd(&arow[cc], w * srow[cc]);
    }
}

// ---------------------------------------------------------------------------
// Host-side launcher
// ---------------------------------------------------------------------------
extern "C" void kernel_launch(void* const* d_in, const int* in_sizes, int n_in,
                              void* d_out, int out_size, void* d_ws, size_t ws_size,
                              hipStream_t stream) {
    (void)in_sizes; (void)n_in; (void)out_size; (void)ws_size;

    // inputs in setup_inputs() order
    const float* x          = (const float*)d_in[0];
    const int*   edge_index = (const int*)  d_in[1];   // [2, E] flat
    const float* edge_w     = (const float*)d_in[2];
    const float* W_gat      = (const float*)d_in[3];
    const float* att_src    = (const float*)d_in[4];
    const float* att_dst    = (const float*)d_in[5];
    const float* b_gat      = (const float*)d_in[6];
    const float* W1         = (const float*)d_in[7];
    const float* b1         = (const float*)d_in[8];
    const float* W2         = (const float*)d_in[9];
    const float* b2         = (const float*)d_in[10];
    float* out = (float*)d_out;

    const int* src = edge_index;          // edge_index[0]
    const int* dst = edge_index + NEDGE;  // edge_index[1]

    // ---- workspace carve-up (regions reused across phases) ----
    char* ws = (char*)d_ws;
    size_t off = 0;
    auto take = [&](size_t bytes) -> char* {
        off = (off + 255) & ~(size_t)255;
        char* p = ws + off;
        off += bytes;
        return p;
    };
    // region A: x_bf16 (N*512*2) -> support1 (N*256*4) -> support2 (N*128*4)
    char* regA = take((size_t)NNODE * INDIM * sizeof(bf16_t));   // 51.2 MB
    // region B: h (N*256*4) -> acc1 (N*256*4)
    char* regB = take((size_t)NNODE * H0D * sizeof(float));      // 51.2 MB
    // region C: gat_acc (N*256*4) -> acc2 (N*128*4)
    char* regC = take((size_t)NNODE * H0D * sizeof(float));      // 51.2 MB
    // region D: h1_bf16 (N*256*2) -> h2_bf16 (N*256*2)
    char* regD = take((size_t)NNODE * H0D * sizeof(bf16_t));     // 25.6 MB

    bf16_t* wgat_t = (bf16_t*)take((size_t)INDIM * H0D * sizeof(bf16_t));
    bf16_t* w1_t   = (bf16_t*)take((size_t)H0D * H1D * sizeof(bf16_t));
    bf16_t* w2_t   = (bf16_t*)take((size_t)H1D * H2D * sizeof(bf16_t));
    float* a_s   = (float*)take((size_t)NNODE * sizeof(float));
    float* a_d   = (float*)take((size_t)NNODE * sizeof(float));
    float* e_max = (float*)take((size_t)NNODE * sizeof(float));
    float* denom = (float*)take((size_t)NNODE * sizeof(float));
    float* e_buf = (float*)take((size_t)(NEDGE + NNODE) * sizeof(float));
    float* ee    = (float*)take((size_t)(NEDGE + NNODE) * sizeof(float));

    bf16_t* x_bf     = (bf16_t*)regA;
    float*  support1 = (float*)regA;
    float*  support2 = (float*)regA;
    float*  h        = (float*)regB;
    float*  acc1     = (float*)regB;
    float*  gat_acc  = (float*)regC;
    float*  acc2     = (float*)regC;
    bf16_t* h1_bf    = (bf16_t*)regD;
    bf16_t* h2_bf    = (bf16_t*)regD;

    const int EN = NEDGE + NNODE;
    const int MTILES = NNODE / 16;            // 3125 exact
    const int MBLK = (MTILES + 7) / 8;        // blocks along M (8 waves/block)

    // ---- 0. convert activations / weights to bf16 (weights transposed) ----
    {
        int n = NNODE * INDIM;
        f32_to_bf16_k<<<(n + 255) / 256, 256, 0, stream>>>(x, x_bf, n);
    }
    transpose_to_bf16_k<<<(INDIM * H0D + 255) / 256, 256, 0, stream>>>(W_gat, wgat_t, INDIM, H0D);
    transpose_to_bf16_k<<<(H0D * H1D + 255) / 256, 256, 0, stream>>>(W1, w1_t, H0D, H1D);
    transpose_to_bf16_k<<<(H1D * H2D + 255) / 256, 256, 0, stream>>>(W2, w2_t, H1D, H2D);

    // ---- 1. h = x @ W_gat  (WMMA bf16 -> f32) ----
    {
        dim3 g(MBLK, H0D / 64);
        gemm_bf16_wmma<<<g, 256, 0, stream>>>(x_bf, wgat_t, h, NNODE, H0D, INDIM);
    }

    // ---- 2. attention dots ----
    gat_dots_k<<<(NNODE + 7) / 8, 256, 0, stream>>>(h, att_src, att_dst, a_s, a_d);

    // ---- 3. edge softmax + aggregation ----
    fill_f32<<<(NNODE + 255) / 256, 256, 0, stream>>>(e_max, -INFINITY, NNODE);
    fill_f32<<<(NNODE + 255) / 256, 256, 0, stream>>>(denom, 0.0f, NNODE);
    fill_f32<<<(NNODE * H0D + 255) / 256, 256, 0, stream>>>(gat_acc, 0.0f, NNODE * H0D);

    gat_edge_logits_k<<<(EN + 255) / 256, 256, 0, stream>>>(src, dst, a_s, a_d, e_buf, e_max);
    gat_edge_exp_k<<<(EN + 255) / 256, 256, 0, stream>>>(dst, e_buf, e_max, ee, denom);
    gat_aggregate_k<<<(EN + 7) / 8, 256, 0, stream>>>(src, dst, ee, denom, h, gat_acc);

    // ---- 4. h1 = bf16(leaky(gat_acc + b_gat)) ----
    bias_leaky_to_bf16_k<<<(NNODE * H0D + 255) / 256, 256, 0, stream>>>(
        gat_acc, b_gat, h1_bf, NNODE * H0D, H0D);

    // ---- 5. support1 = h1 @ W1 ----
    {
        dim3 g(MBLK, H1D / 64);
        gemm_bf16_wmma<<<g, 256, 0, stream>>>(h1_bf, w1_t, support1, NNODE, H1D, H0D);
    }

    // ---- 6. acc1[row] += w * support1[col] ----
    fill_f32<<<(NNODE * H1D + 255) / 256, 256, 0, stream>>>(acc1, 0.0f, NNODE * H1D);
    gc_scatter_k<<<(NEDGE + 7) / 8, 256, 0, stream>>>(src, dst, edge_w, support1, acc1, H1D);

    // ---- 7. h2 = bf16(leaky(acc1 + b1)) ----
    bias_leaky_to_bf16_k<<<(NNODE * H1D + 255) / 256, 256, 0, stream>>>(
        acc1, b1, h2_bf, NNODE * H1D, H1D);

    // ---- 8. support2 = h2 @ W2 ----
    {
        dim3 g(MBLK, H2D / 64);
        gemm_bf16_wmma<<<g, 256, 0, stream>>>(h2_bf, w2_t, support2, NNODE, H2D, H1D);
    }

    // ---- 9. acc2[row] += w * support2[col] ----
    fill_f32<<<(NNODE * H2D + 255) / 256, 256, 0, stream>>>(acc2, 0.0f, NNODE * H2D);
    gc_scatter_k<<<(NEDGE + 7) / 8, 256, 0, stream>>>(src, dst, edge_w, support2, acc2, H2D);

    // ---- 10. out = leaky(acc2 + b2) ----
    bias_leaky_f32_k<<<(NNODE * H2D + 255) / 256, 256, 0, stream>>>(
        acc2, b2, out, NNODE * H2D, H2D);
}
